// GRAS_5334349381680
// MI455X (gfx1250) — matrix operand order repaired and checked by reading, not compiled, not run on hardware
//
#include <hip/hip_runtime.h>
#include <hip/hip_bf16.h>

// Problem sizes (fixed by the reference)
#define T_DIM 8192
#define D_DIM 2048
#define R_DIM 2048
#define LC    128          // scan chunk length
#define NC    (T_DIM / LC) // 64 chunks

// GEMM tiling
#define BM 128
#define BN 64
#define BK 32
#define WAVES_M 4
#define WAVES_N 2

typedef __attribute__((ext_vector_type(16))) __bf16 v16bf;
typedef __attribute__((ext_vector_type(8)))  float  v8f;
typedef __attribute__((ext_vector_type(4)))  unsigned int v4u;
typedef __attribute__((ext_vector_type(8)))  unsigned int v8u;

union FragBf {
    uint4 q[2];
    v16bf v;
};

__device__ __forceinline__ unsigned short f32_to_bf16_rne(float f) {
    unsigned u = __float_as_uint(f);
    u += 0x7fffu + ((u >> 16) & 1u);
    return (unsigned short)(u >> 16);
}

// LDS byte offset (wave-relative) of a __shared__ object: generic AS pointer is
// {aperture_hi, lds_offset32}; truncation yields the DS-instruction address.
__device__ __forceinline__ unsigned lds_off(const void* p) {
    return (unsigned)(unsigned long long)p;
}

// CDNA5 async DMA: global -> LDS, 16 bytes per lane, tracked by ASYNCcnt.
// GVS addressing: mem = SADDR(64b) + VADDR(32b byte offset).
__device__ __forceinline__ void async_load_b128(unsigned lds_byte_off,
                                                const void* base,
                                                unsigned voff_bytes) {
    asm volatile("global_load_async_to_lds_b128 %0, %1, %2"
                 :
                 : "v"(lds_byte_off), "v"(voff_bytes),
                   "s"((unsigned long long)base)
                 : "memory");
}

#define ASYNC_WAIT(N) asm volatile("s_wait_asynccnt " #N ::: "memory")

// CDNA5 Tensor Data Mover: DMA a 2-D tile (tile_rows x BK bf16 elements) of a
// row-major tensor (nrows x D_DIM, stride D_DIM) into LDS, packed contiguously.
// D# groups per CDNA5 ISA; issued via inline asm (toolchain-portable).
__device__ __forceinline__ void tdm_load_2d(unsigned lds_addr,
                                            const void* gbase,
                                            unsigned long long elem_off,
                                            unsigned nrows,
                                            unsigned tile_rows) {
    unsigned long long ga = (unsigned long long)gbase + elem_off * 2ull;
    v4u g0;
    g0[0] = 1u;                                       // count=1 (valid descriptor)
    g0[1] = lds_addr;                                 // lds_addr [63:32]
    g0[2] = (unsigned)ga;                             // global_addr lo
    g0[3] = (unsigned)((ga >> 32) & 0x01FFFFFFull)    // global_addr hi (57-bit)
          | (2u << 30);                               // type=2 ("image")
    v8u g1;
    g1[0] = 0x00010000u;                              // wg_mask=0, data_size=1 (2B)
    g1[1] = ((unsigned)D_DIM & 0xFFFFu) << 16;        // tensor_dim0[15:0] @ [63:48]
    g1[2] = (nrows & 0xFFFFu) << 16;                  // tensor_dim1[15:0] @ [95:80]
    g1[3] = ((unsigned)BK & 0xFFFFu) << 16;           // tile_dim0 @ [127:112]
    g1[4] = tile_rows & 0xFFFFu;                      // tile_dim1 @ [143:128]
    g1[5] = (unsigned)D_DIM;                          // tensor_dim0_stride lo32
    g1[6] = 0u;
    g1[7] = 0u;
    asm volatile("tensor_load_to_lds %0, %1"
                 :
                 : "s"(g0), "s"(g1)
                 : "memory");
}

// ---------------------------------------------------------------------------
// Kernel: f32 -> bf16 conversion (grid-stride)
// ---------------------------------------------------------------------------
__global__ void cvt_bf16_kernel(const float* __restrict__ src,
                                unsigned short* __restrict__ dst, int n) {
    int i = blockIdx.x * blockDim.x + threadIdx.x;
    int stride = gridDim.x * blockDim.x;
    for (; i < n; i += stride) {
        dst[i] = f32_to_bf16_rne(src[i]);
    }
}

// ---------------------------------------------------------------------------
// Kernel: f32 (K x N) -> bf16 transposed (N x K), tiled via LDS.
// ---------------------------------------------------------------------------
__global__ __launch_bounds__(256) void cvt_transpose_kernel(
    const float* __restrict__ src,          // K x N
    unsigned short* __restrict__ dst,       // N x K
    int K, int N)
{
    __shared__ float tile[32][33];
    const int bn = blockIdx.x * 32;   // N tile origin
    const int bk = blockIdx.y * 32;   // K tile origin
    const int tx = threadIdx.x;       // 0..31
    const int ty = threadIdx.y;       // 0..7

    #pragma unroll
    for (int j = 0; j < 32; j += 8)
        tile[ty + j][tx] = src[(size_t)(bk + ty + j) * N + bn + tx];
    __syncthreads();
    #pragma unroll
    for (int j = 0; j < 32; j += 8)
        dst[(size_t)(bn + ty + j) * K + bk + tx] = f32_to_bf16_rne(tile[tx][ty + j]);
}

// ---------------------------------------------------------------------------
// Kernel: fused gate GEMM.  a = sigmoid(x@Wa), b = (x@Wb) * (1-a)
// X: T x D (bf16), WaT/WbT: R x D (bf16, pre-transposed), outA/outB: T x R f32
// A tile staged via TDM (tensor_load_to_lds), B tiles via async global->LDS;
// double-buffered with TENSORcnt/ASYNCcnt waits.
// ---------------------------------------------------------------------------
__global__ __launch_bounds__(256) void gemm_gate_kernel(
    const unsigned short* __restrict__ X,
    const unsigned short* __restrict__ WaT,
    const unsigned short* __restrict__ WbT,
    float* __restrict__ outA,
    float* __restrict__ outB)
{
    __shared__ __align__(16) unsigned short As[2][BM * BK];
    __shared__ __align__(16) unsigned short Ba[2][BN * BK];
    __shared__ __align__(16) unsigned short Bb[2][BN * BK];

    const int tid  = threadIdx.x;
    const int lane = tid & 31;
    const int wave = tid >> 5;
    const int wm   = wave % WAVES_M;   // 0..3
    const int wn   = wave / WAVES_M;   // 0..1
    const int m0   = blockIdx.y * BM;
    const int n0   = blockIdx.x * BN;
    const int lrow = lane & 15;
    const int lhi  = lane >> 4;

    v8f accP[2][2] = {};
    v8f accQ[2][2] = {};

    // Per-thread async-staging coordinates for B tiles (constant across K-steps)
    const int bn_r = tid >> 2;               // n 0..63
    const int bk_c = (tid & 3) * 8;          // k col 0,8,16,24

    auto stage = [&](int buf, int k0) {
        // B tiles 64x32 bf16 (N-major, pre-transposed weights): 1 async each
        unsigned bvoff = ((unsigned)(n0 + bn_r) * D_DIM + (unsigned)(k0 + bk_c)) * 2u;
        async_load_b128(lds_off(&Ba[buf][bn_r * BK + bk_c]), WaT, bvoff);
        async_load_b128(lds_off(&Bb[buf][bn_r * BK + bk_c]), WbT, bvoff);
        // A tile 128x32 bf16: single TDM descriptor, issued by wave 0 only
        if (wave == 0)
            tdm_load_2d(lds_off(&As[buf][0]), X,
                        (unsigned long long)m0 * D_DIM + (unsigned)k0,
                        T_DIM, BM);
    };

    const int NK = D_DIM / BK;
    stage(0, 0);

    for (int ki = 0; ki < NK; ++ki) {
        const int cur = ki & 1;
        if (ki + 1 < NK) {
            stage(cur ^ 1, (ki + 1) * BK);
            ASYNC_WAIT(2);                         // current buffer's 2 async done
            __builtin_amdgcn_s_wait_tensorcnt(1);  // current buffer's TDM done
        } else {
            ASYNC_WAIT(0);
            __builtin_amdgcn_s_wait_tensorcnt(0);
        }
        __syncthreads();

        FragBf afr[2];
        #pragma unroll
        for (int fm = 0; fm < 2; ++fm) {
            int row = wm * 32 + fm * 16 + lrow;
            int kb  = lhi * 16;            // lanes 0-15: K 0..15; 16-31: K 16..31
            afr[fm].q[0] = *(const uint4*)&As[cur][row * BK + kb];
            afr[fm].q[1] = *(const uint4*)&As[cur][row * BK + kb + 8];
        }
        FragBf bfa[2], bfb[2];
        #pragma unroll
        for (int fn = 0; fn < 2; ++fn) {
            int n  = wn * 32 + fn * 16 + lrow;
            int kb = lhi * 16;             // lanes 0-15: K 0..15; 16-31: K 16..31
            bfa[fn].q[0] = *(const uint4*)&Ba[cur][n * BK + kb];
            bfa[fn].q[1] = *(const uint4*)&Ba[cur][n * BK + kb + 8];
            bfb[fn].q[0] = *(const uint4*)&Bb[cur][n * BK + kb];
            bfb[fn].q[1] = *(const uint4*)&Bb[cur][n * BK + kb + 8];
        }

        #pragma unroll
        for (int fm = 0; fm < 2; ++fm)
            #pragma unroll
            for (int fn = 0; fn < 2; ++fn) {
                accP[fm][fn] = __builtin_amdgcn_wmma_f32_16x16x32_bf16(
                    false, afr[fm].v, false, bfa[fn].v, (short)0, accP[fm][fn], false, false);
                accQ[fm][fn] = __builtin_amdgcn_wmma_f32_16x16x32_bf16(
                    false, afr[fm].v, false, bfb[fn].v, (short)0, accQ[fm][fn], false, false);
            }
        __syncthreads();
    }

    // epilogue: a = sigmoid(p); b = q*(1-a)
    #pragma unroll
    for (int fm = 0; fm < 2; ++fm)
        #pragma unroll
        for (int fn = 0; fn < 2; ++fn)
            #pragma unroll
            for (int j = 0; j < 8; ++j) {
                int mm = m0 + wm * 32 + fm * 16 + 8 * lhi + j;
                int nn = n0 + wn * 32 + fn * 16 + lrow;
                float p = accP[fm][fn][j];
                float q = accQ[fm][fn][j];
                float a = __builtin_amdgcn_rcpf(1.0f + __expf(-p));
                outA[(size_t)mm * R_DIM + nn] = a;
                outB[(size_t)mm * R_DIM + nn] = q * (1.0f - a);
            }
}

// ---------------------------------------------------------------------------
// Kernel: per-chunk aggregate of the recurrence (A,B) composition
// ---------------------------------------------------------------------------
__global__ void chunk_agg_kernel(const float* __restrict__ A,
                                 const float* __restrict__ B,
                                 float* __restrict__ aggA,
                                 float* __restrict__ aggB)
{
    int idx = blockIdx.x * blockDim.x + threadIdx.x;   // NC*R threads
    int r = idx % R_DIM;
    int c = idx / R_DIM;
    size_t base = (size_t)c * LC * R_DIM + r;
    float pA = 1.0f, pB = 0.0f;
    for (int t = 0; t < LC; ++t) {
        float at = A[base + (size_t)t * R_DIM];
        float bt = B[base + (size_t)t * R_DIM];
        pA = at * pA;
        pB = at * pB + bt;
    }
    aggA[idx] = pA;
    aggB[idx] = pB;
}

// ---------------------------------------------------------------------------
// Kernel: sequential carry scan over the NC chunk aggregates
// ---------------------------------------------------------------------------
__global__ void scan_carry_kernel(const float* __restrict__ aggA,
                                  const float* __restrict__ aggB,
                                  const float* __restrict__ h0,
                                  float* __restrict__ carry)
{
    int r = blockIdx.x * blockDim.x + threadIdx.x;     // R threads
    float h = h0[r];
    for (int c = 0; c < NC; ++c) {
        carry[(size_t)c * R_DIM + r] = h;
        h = aggA[(size_t)c * R_DIM + r] * h + aggB[(size_t)c * R_DIM + r];
    }
}

// ---------------------------------------------------------------------------
// Kernel: apply scan within each chunk; emit next_h (f32) + tanh(h) (bf16)
// ---------------------------------------------------------------------------
__global__ void apply_scan_kernel(const float* __restrict__ A,
                                  const float* __restrict__ B,
                                  const float* __restrict__ carry,
                                  float* __restrict__ Hout,
                                  unsigned short* __restrict__ TH)
{
    int idx = blockIdx.x * blockDim.x + threadIdx.x;
    int r = idx % R_DIM;
    int c = idx / R_DIM;
    float h = carry[(size_t)c * R_DIM + r];
    size_t base = (size_t)c * LC * R_DIM + r;
    for (int t = 0; t < LC; ++t) {
        float at = A[base + (size_t)t * R_DIM];
        float bt = B[base + (size_t)t * R_DIM];
        h = at * h + bt;
        Hout[base + (size_t)t * R_DIM] = h;
        TH[base + (size_t)t * R_DIM] = f32_to_bf16_rne(tanhf(h));
    }
}

// ---------------------------------------------------------------------------
// Kernel: fused output GEMM  y = tanh_h@W_out + x@W_skip
// Flattened 2-phase K loop; A via TDM, B via async; double-buffered.
// ---------------------------------------------------------------------------
__global__ __launch_bounds__(256) void gemm_out_kernel(
    const unsigned short* __restrict__ TH,      // T x R bf16
    const unsigned short* __restrict__ WoutT,   // D x R bf16 (transposed: N x K)
    const unsigned short* __restrict__ XH,      // T x D bf16
    const unsigned short* __restrict__ WskipT,  // D x D bf16 (transposed: N x K)
    float* __restrict__ Y)
{
    __shared__ __align__(16) unsigned short As[2][BM * BK];
    __shared__ __align__(16) unsigned short Bs[2][BN * BK];

    const int tid  = threadIdx.x;
    const int lane = tid & 31;
    const int wave = tid >> 5;
    const int wm   = wave % WAVES_M;
    const int wn   = wave / WAVES_M;
    const int m0   = blockIdx.y * BM;
    const int n0   = blockIdx.x * BN;
    const int lrow = lane & 15;
    const int lhi  = lane >> 4;

    v8f acc[2][2] = {};

    const int bn_r = tid >> 2;
    const int bk_c = (tid & 3) * 8;

    const unsigned short* Aptr[2] = { TH, XH };
    const unsigned short* Bptr[2] = { WoutT, WskipT };

    const int NK  = D_DIM / BK;      // per phase
    const int NK2 = 2 * NK;

    auto stage = [&](int buf, int s) {
        const unsigned short* Am = Aptr[s >> 6];
        const unsigned short* Bm = Bptr[s >> 6];
        int k0 = (s & (NK - 1)) * BK;
        async_load_b128(lds_off(&Bs[buf][bn_r * BK + bk_c]), Bm,
                        ((unsigned)(n0 + bn_r) * D_DIM + (unsigned)(k0 + bk_c)) * 2u);
        if (wave == 0)
            tdm_load_2d(lds_off(&As[buf][0]), Am,
                        (unsigned long long)m0 * D_DIM + (unsigned)k0,
                        T_DIM, BM);
    };

    stage(0, 0);

    for (int s = 0; s < NK2; ++s) {
        const int cur = s & 1;
        if (s + 1 < NK2) {
            stage(cur ^ 1, s + 1);
            ASYNC_WAIT(1);
            __builtin_amdgcn_s_wait_tensorcnt(1);
        } else {
            ASYNC_WAIT(0);
            __builtin_amdgcn_s_wait_tensorcnt(0);
        }
        __syncthreads();

        FragBf afr[2], bfr[2];
        #pragma unroll
        for (int fm = 0; fm < 2; ++fm) {
            int row = wm * 32 + fm * 16 + lrow;
            int kb  = lhi * 16;
            afr[fm].q[0] = *(const uint4*)&As[cur][row * BK + kb];
            afr[fm].q[1] = *(const uint4*)&As[cur][row * BK + kb + 8];
        }
        #pragma unroll
        for (int fn = 0; fn < 2; ++fn) {
            int n  = wn * 32 + fn * 16 + lrow;
            int kb = lhi * 16;
            bfr[fn].q[0] = *(const uint4*)&Bs[cur][n * BK + kb];
            bfr[fn].q[1] = *(const uint4*)&Bs[cur][n * BK + kb + 8];
        }
        #pragma unroll
        for (int fm = 0; fm < 2; ++fm)
            #pragma unroll
            for (int fn = 0; fn < 2; ++fn)
                acc[fm][fn] = __builtin_amdgcn_wmma_f32_16x16x32_bf16(
                    false, afr[fm].v, false, bfr[fn].v, (short)0, acc[fm][fn], false, false);
        __syncthreads();
    }

    #pragma unroll
    for (int fm = 0; fm < 2; ++fm)
        #pragma unroll
        for (int fn = 0; fn < 2; ++fn)
            #pragma unroll
            for (int j = 0; j < 8; ++j) {
                int mm = m0 + wm * 32 + fm * 16 + 8 * lhi + j;
                int nn = n0 + wn * 32 + fn * 16 + lrow;
                Y[(size_t)mm * D_DIM + nn] = acc[fm][fn][j];
            }
}

// ---------------------------------------------------------------------------
extern "C" void kernel_launch(void* const* d_in, const int* in_sizes, int n_in,
                              void* d_out, int out_size, void* d_ws, size_t ws_size,
                              hipStream_t stream) {
    (void)in_sizes; (void)n_in; (void)out_size; (void)ws_size;

    const float* x      = (const float*)d_in[0];   // T x D
    const float* h0     = (const float*)d_in[1];   // R
    const float* W_a    = (const float*)d_in[2];   // D x R
    const float* W_b    = (const float*)d_in[3];   // D x R
    const float* W_out  = (const float*)d_in[4];   // R x D
    const float* W_skip = (const float*)d_in[5];   // D x D

    float* next_h = (float*)d_out;                         // T x R
    float* y      = next_h + (size_t)T_DIM * R_DIM;        // T x D

    // workspace carve-up
    float* ws_a = (float*)d_ws;                                    // T*R f32
    float* ws_b = ws_a + (size_t)T_DIM * R_DIM;                    // T*R f32
    unsigned short* xh     = (unsigned short*)(ws_b + (size_t)T_DIM * R_DIM); // T*D bf16
    unsigned short* waT    = xh     + (size_t)T_DIM * D_DIM;       // R x D bf16
    unsigned short* wbT    = waT    + (size_t)D_DIM * R_DIM;       // R x D
    unsigned short* woutT  = wbT    + (size_t)D_DIM * R_DIM;       // D x R
    unsigned short* wskipT = woutT  + (size_t)R_DIM * D_DIM;       // D x D
    unsigned short* th_h   = wskipT + (size_t)D_DIM * D_DIM;       // T*R bf16
    float* aggA  = (float*)(th_h + (size_t)T_DIM * R_DIM);         // NC*R
    float* aggB  = aggA + (size_t)NC * R_DIM;
    float* carry = aggB + (size_t)NC * R_DIM;

    // 1) down-convert x; transpose+convert weights (N-major for B staging)
    cvt_bf16_kernel<<<2048, 256, 0, stream>>>(x, xh, T_DIM * D_DIM);
    dim3 tgrid(R_DIM / 32, D_DIM / 32);
    dim3 tblk(32, 8);
    cvt_transpose_kernel<<<tgrid, tblk, 0, stream>>>(W_a,    waT,    D_DIM, R_DIM);
    cvt_transpose_kernel<<<tgrid, tblk, 0, stream>>>(W_b,    wbT,    D_DIM, R_DIM);
    cvt_transpose_kernel<<<dim3(D_DIM / 32, R_DIM / 32), tblk, 0, stream>>>(W_out,  woutT,  R_DIM, D_DIM);
    cvt_transpose_kernel<<<dim3(D_DIM / 32, D_DIM / 32), tblk, 0, stream>>>(W_skip, wskipT, D_DIM, D_DIM);

    // 2) fused gate GEMM (a, b)
    dim3 ggrid(R_DIM / BN, T_DIM / BM);
    gemm_gate_kernel<<<ggrid, 256, 0, stream>>>(xh, waT, wbT, ws_a, ws_b);

    // 3-5) chunked linear-recurrence scan
    chunk_agg_kernel<<<(NC * R_DIM) / 256, 256, 0, stream>>>(ws_a, ws_b, aggA, aggB);
    scan_carry_kernel<<<R_DIM / 256, 256, 0, stream>>>(aggA, aggB, h0, carry);
    apply_scan_kernel<<<(NC * R_DIM) / 256, 256, 0, stream>>>(ws_a, ws_b, carry, next_h, th_h);

    // 6) fused output GEMM: y = tanh_h@W_out + x@W_skip
    dim3 ogrid(D_DIM / BN, T_DIM / BM);
    gemm_out_kernel<<<ogrid, 256, 0, stream>>>(th_h, woutT, xh, wskipT, y);
}
